// GraphConvolution_23888608100646
// MI455X (gfx1250) — compile-verified, matrix operand order checked by reading, not run on hardware
//
#include <hip/hip_runtime.h>
#include <hip/hip_fp16.h>

#define D 256

typedef __attribute__((ext_vector_type(16))) _Float16 v16h;
typedef __attribute__((ext_vector_type(8)))  _Float16 v8h;
typedef __attribute__((ext_vector_type(8)))  float    v8f;

// ---------------- f32 -> f16 conversion of x ----------------
__global__ void cvt_x_kernel(const float* __restrict__ x, _Float16* __restrict__ x16, int n) {
    int i = blockIdx.x * 256 + threadIdx.x;
    if (i < n) x16[i] = (_Float16)x[i];
}

// ------- transpose + cast the 3 weight matrices: wt[mat][n][k] = w[k][n] -------
__global__ void cvt_wt_kernel(const float* __restrict__ wl, const float* __restrict__ wh,
                              const float* __restrict__ wm, _Float16* __restrict__ wt) {
    int i = blockIdx.x * 256 + threadIdx.x;          // 0 .. 3*256*256-1
    if (i >= 3 * D * D) return;
    int mat = i >> 16;
    int idx = i & 0xFFFF;
    int n = idx >> 8;
    int k = idx & 0xFF;
    const float* w = (mat == 0) ? wl : ((mat == 1) ? wh : wm);
    wt[i] = (_Float16)w[k * D + n];
}

// ---------------- WMMA GEMM: H[mat] = x @ W[mat]  (relu fused for mat==2) ------
// block = 128 threads = 4 waves; each wave computes one 16x16 tile of H.
// grid = (N/16, 4, 3): x = row tile, y*4+wave = col tile (16 total), z = matrix.
__global__ __launch_bounds__(128) void gemm_wmma_kernel(
    const _Float16* __restrict__ x16,       // [N][256] row-major f16
    const _Float16* __restrict__ wt,        // [3][256 n][256 k] (transposed) f16
    float* __restrict__ H, int N) {
    const int wave = threadIdx.x >> 5;
    const int lane = threadIdx.x & 31;
    const int half = lane >> 4;             // 0: lanes 0-15, 1: lanes 16-31
    const int mn   = lane & 15;             // M index for A, N index for B/D
    const int rowBase = blockIdx.x * 16;
    const int colTile = blockIdx.y * 4 + wave;      // 0..15
    const int mat = blockIdx.z;

    const _Float16* aRow = x16 + (size_t)(rowBase + mn) * D;
    const _Float16* bRow = wt + (size_t)mat * D * D + (size_t)(colTile * 16 + mn) * D;

    v8f c = {};
    #pragma unroll
    for (int kk = 0; kk < D; kk += 32) {
        const int k0 = kk + half * 8;
        // A 16x32 f16 layout: elems 0..7 -> K = k0..k0+7 ; elems 8..15 -> K = k0+16..k0+23
        v8h a0 = *(const v8h*)(aRow + k0);
        v8h a1 = *(const v8h*)(aRow + k0 + 16);
        v8h b0 = *(const v8h*)(bRow + k0);
        v8h b1 = *(const v8h*)(bRow + k0 + 16);
        v16h a = __builtin_shufflevector(a0, a1, 0,1,2,3,4,5,6,7,8,9,10,11,12,13,14,15);
        v16h b = __builtin_shufflevector(b0, b1, 0,1,2,3,4,5,6,7,8,9,10,11,12,13,14,15);
        c = __builtin_amdgcn_wmma_f32_16x16x32_f16(false, a, false, b, (short)0, c, false, false);
    }

    // D layout: VGPR r holds (M = half*8 + r, N = mn)
    float* outp = H + (size_t)mat * N * D + (size_t)(rowBase + half * 8) * D + colTile * 16 + mn;
    const bool isMlp = (mat == 2);
    #pragma unroll
    for (int r = 0; r < 8; ++r) {
        float v = c[r];
        if (isMlp) v = fmaxf(v, 0.0f);
        outp[(size_t)r * D] = v;
    }
}

// ---------------- SpMM (both low & high): block = one edge, thread = one dim ---
__global__ __launch_bounds__(256) void spmm_kernel(
    const int* __restrict__ row, const int* __restrict__ col,
    const float* __restrict__ val_low, const float* __restrict__ val_high,
    const float* __restrict__ HL, const float* __restrict__ HH,
    float* __restrict__ accL, float* __restrict__ accH) {
    const int e = blockIdx.x;
    const int d = threadIdx.x;
    const int r = row[e];
    const int c = col[e];
    const float vl = val_low[e];
    const float vh = val_high[e];
    const size_t co = (size_t)c * D + d;
    const size_t ro = (size_t)r * D + d;
    unsafeAtomicAdd(&accL[ro], vl * HL[co]);
    unsafeAtomicAdd(&accH[ro], vh * HH[co]);
}

// ---------------- attention + combine: one wave per node, 8 dims per lane ------
__global__ __launch_bounds__(256) void fuse_kernel(
    const float* __restrict__ accL, const float* __restrict__ accH,
    const float* __restrict__ Hm,
    const float* __restrict__ avL, const float* __restrict__ avH,
    const float* __restrict__ avM, const float* __restrict__ av,
    float* __restrict__ out) {
    const int wave = threadIdx.x >> 5;
    const int lane = threadIdx.x & 31;
    const int node = blockIdx.x * 8 + wave;
    const size_t base = (size_t)node * D + lane * 8;

    float l[8], h[8], m[8];
    float d0 = 0.f, d1 = 0.f, d2 = 0.f;
    #pragma unroll
    for (int j = 0; j < 8; ++j) {
        const int dd = lane * 8 + j;
        l[j] = fmaxf(accL[base + j], 0.f);      // relu(spmm_low)
        h[j] = fmaxf(accH[base + j], 0.f);      // relu(spmm_high)
        m[j] = Hm[base + j];                    // relu already fused in GEMM
        d0 += l[j] * avL[dd];
        d1 += h[j] * avH[dd];
        d2 += m[j] * avM[dd];
    }
    // wave32 butterfly reduction -> every lane holds the full dot products
    #pragma unroll
    for (int off = 16; off >= 1; off >>= 1) {
        d0 += __shfl_xor(d0, off, 32);
        d1 += __shfl_xor(d1, off, 32);
        d2 += __shfl_xor(d2, off, 32);
    }
    const float s0 = 1.f / (1.f + __expf(-d0));
    const float s1 = 1.f / (1.f + __expf(-d1));
    const float s2 = 1.f / (1.f + __expf(-d2));
    const float g0 = (s0 * av[0] + s1 * av[3] + s2 * av[6]) * (1.f / 3.f);
    const float g1 = (s0 * av[1] + s1 * av[4] + s2 * av[7]) * (1.f / 3.f);
    const float g2 = (s0 * av[2] + s1 * av[5] + s2 * av[8]) * (1.f / 3.f);
    const float gm = fmaxf(g0, fmaxf(g1, g2));
    const float e0 = __expf(g0 - gm), e1 = __expf(g1 - gm), e2 = __expf(g2 - gm);
    const float inv = 3.f / (e0 + e1 + e2);     // folds the final 3.0x scale
    const float a0 = e0 * inv, a1 = e1 * inv, a2 = e2 * inv;
    #pragma unroll
    for (int j = 0; j < 8; ++j)
        out[base + j] = a0 * l[j] + a1 * h[j] + a2 * m[j];
}

extern "C" void kernel_launch(void* const* d_in, const int* in_sizes, int n_in,
                              void* d_out, int out_size, void* d_ws, size_t ws_size,
                              hipStream_t stream) {
    const float* x        = (const float*)d_in[0];
    const float* val_low  = (const float*)d_in[1];
    const float* val_high = (const float*)d_in[2];
    const float* w_low    = (const float*)d_in[3];
    const float* w_high   = (const float*)d_in[4];
    const float* w_mlp    = (const float*)d_in[5];
    const float* av_low   = (const float*)d_in[6];
    const float* av_high  = (const float*)d_in[7];
    const float* av_mlp   = (const float*)d_in[8];
    const float* av       = (const float*)d_in[9];
    const int*   row      = (const int*)d_in[10];
    const int*   col      = (const int*)d_in[11];

    const int N = in_sizes[0] / D;     // 50000
    const int E = in_sizes[1];         // 800000

    // workspace layout (all offsets 256B aligned)
    char* ws = (char*)d_ws;
    size_t off = 0;
    _Float16* x16 = (_Float16*)(ws + off); off += (size_t)N * D * sizeof(_Float16);  // 25.6 MB
    _Float16* wt  = (_Float16*)(ws + off); off += (size_t)3 * D * D * sizeof(_Float16);
    float* H      = (float*)(ws + off);    off += (size_t)3 * N * D * sizeof(float); // 153.6 MB
    float* accL   = (float*)(ws + off);    off += (size_t)N * D * sizeof(float);
    float* accH   = (float*)(ws + off);    off += (size_t)N * D * sizeof(float);

    // zero the two scatter-accumulators (contiguous)
    hipMemsetAsync(accL, 0, 2 * (size_t)N * D * sizeof(float), stream);

    cvt_x_kernel<<<(N * D + 255) / 256, 256, 0, stream>>>(x, x16, N * D);
    cvt_wt_kernel<<<(3 * D * D + 255) / 256, 256, 0, stream>>>(w_low, w_high, w_mlp, wt);

    gemm_wmma_kernel<<<dim3(N / 16, 4, 3), 128, 0, stream>>>(x16, wt, H, N);

    spmm_kernel<<<E, 256, 0, stream>>>(row, col, val_low, val_high,
                                       H, H + (size_t)N * D, accL, accH);

    fuse_kernel<<<N / 8, 256, 0, stream>>>(accL, accH, H + 2 * (size_t)N * D,
                                           av_low, av_high, av_mlp, av, (float*)d_out);
}
